// DeepSeekSparseAttention_20675972563683
// MI455X (gfx1250) — matrix operand order repaired and checked by reading, not compiled, not run on hardware
//
#include <hip/hip_runtime.h>
#include <hip/hip_bf16.h>

#define BATCH   2
#define TSEQ    4096
#define DMODEL  768
#define NHEADS  12
#define HDIM    64
#define HI_N    4
#define TOPKN   256
#define NTOK    (BATCH * TSEQ)

typedef __attribute__((ext_vector_type(16))) __bf16 v16bf;
typedef __attribute__((ext_vector_type(8)))  float  v8f;
typedef unsigned short bf16_t;

union FragU { v16bf v; uint4 q[2]; };

__device__ __forceinline__ bf16_t f2b(float f) {
  unsigned u = __float_as_uint(f);
  unsigned r = u + 0x7fffu + ((u >> 16) & 1u);
  return (bf16_t)(r >> 16);
}

// A fragment: 16x32 bf16, row-major source (ld elems/row).
// lane: m = lane%16 ; VGPR0..3 -> kb..kb+7, VGPR4..7 -> kb+16..kb+23, kb = k0 + 8*(lane>=16)
__device__ __forceinline__ v16bf load_a(const bf16_t* base, int ld, int m0, int k0, int lane) {
  int m  = m0 + (lane & 15);
  int kb = k0 + ((lane >> 4) << 3);
  FragU f;
  f.q[0] = *(const uint4*)(base + (size_t)m * ld + kb);
  f.q[1] = *(const uint4*)(base + (size_t)m * ld + kb + 16);
  return f.v;
}

// B fragment: 32x16 bf16 where B[k][n] = W[n][k], W row-major (ld elems/row).
// lane: n = lane%16 ; VGPR0..3 -> kb..kb+7, VGPR4..7 -> kb+16..kb+23, kb = k0 + 8*(lane>=16)
__device__ __forceinline__ v16bf load_b(const bf16_t* base, int ld, int n0, int k0, int lane) {
  int n  = n0 + (lane & 15);
  int kb = k0 + ((lane >> 4) << 3);
  FragU f;
  f.q[0] = *(const uint4*)(base + (size_t)n * ld + kb);
  f.q[1] = *(const uint4*)(base + (size_t)n * ld + kb + 16);
  return f.v;
}

__device__ __forceinline__ unsigned mkey16(bf16_t h) {
  unsigned u = h;
  return (u & 0x8000u) ? (0xffffu & ~u) : (u | 0x8000u);
}

// ---------------- elementwise converts ----------------
__global__ __launch_bounds__(256)
void cvt_f2b(const float* __restrict__ s, bf16_t* __restrict__ d, int nElem) {
  int i = blockIdx.x * 256 + threadIdx.x;
  if (i < nElem) d[i] = f2b(s[i]);
}

__global__ __launch_bounds__(256)
void cvt_v(const float* __restrict__ kv, bf16_t* __restrict__ v, int nElem) {
  int i = blockIdx.x * 256 + threadIdx.x;
  if (i < nElem) {
    int n = i >> 6, d = i & 63;
    v[i] = f2b(kv[(size_t)n * 128 + 64 + d]);
  }
}

// ---------------- generic bf16 GEMM: out = A(MxK) * W(NxK)^T + bias ----------------
// 128-thread block = 4 waves; each wave owns a 16x64 output strip:
// one A fragment is reused across 4 B fragments per k-step (4x A-traffic reduction,
// 4 WMMAs issued back-to-back per A load to keep the matrix pipe busy).
__global__ __launch_bounds__(128)
void gemm_bf16(const bf16_t* __restrict__ A, const bf16_t* __restrict__ W,
               const float* __restrict__ bias, float* outF, bf16_t* outB,
               int M, int N, int K) {
  int lane = threadIdx.x & 31;
  int wave = threadIdx.x >> 5;
  int nStrips = N >> 6;                    // 64-wide strips
  int tile = blockIdx.x * 4 + wave;        // total strips multiple of 4 for all launches
  int m0 = (tile / nStrips) << 4;
  int n0 = (tile % nStrips) << 6;
  v8f c0 = {}, c1 = {}, c2 = {}, c3 = {};
  for (int k0 = 0; k0 < K; k0 += 32) {
    // prefetch two k-steps ahead (k0+64) (lowered to global_prefetch_b8; speculative, OOB-safe)
    __builtin_prefetch(A + (size_t)(m0 + (lane & 15)) * K + k0 + 64, 0, 1);
    __builtin_prefetch(W + (size_t)(n0 + (lane & 15)) * K + k0 + 64, 0, 1);
    v16bf a  = load_a(A, K, m0, k0, lane);
    v16bf b0 = load_b(W, K, n0,      k0, lane);
    v16bf b1 = load_b(W, K, n0 + 16, k0, lane);
    v16bf b2 = load_b(W, K, n0 + 32, k0, lane);
    v16bf b3 = load_b(W, K, n0 + 48, k0, lane);
    c0 = __builtin_amdgcn_wmma_f32_16x16x32_bf16(false, a, false, b0, (short)0, c0, false, false);
    c1 = __builtin_amdgcn_wmma_f32_16x16x32_bf16(false, a, false, b1, (short)0, c1, false, false);
    c2 = __builtin_amdgcn_wmma_f32_16x16x32_bf16(false, a, false, b2, (short)0, c2, false, false);
    c3 = __builtin_amdgcn_wmma_f32_16x16x32_bf16(false, a, false, b3, (short)0, c3, false, false);
  }
  int nn = lane & 15;
  int mb = m0 + ((lane >> 4) << 3);
  v8f acc[4] = {c0, c1, c2, c3};
#pragma unroll
  for (int i = 0; i < 4; ++i) {
    int n = n0 + (i << 4) + nn;
    float bv = bias[n];
#pragma unroll
    for (int r = 0; r < 8; ++r) {
      float v = acc[i][r] + bv;
      size_t o = (size_t)(mb + r) * N + n;
      if (outF) outF[o] = v;
      if (outB) outB[o] = f2b(v);
    }
  }
}

// ---------------- RoPE-2D (fp32 in -> bf16 out) ----------------
__global__ __launch_bounds__(256)
void rope_kernel(const float* __restrict__ src, int ld_src,
                 bf16_t* __restrict__ dst, int ld_dst, int heads,
                 const int* __restrict__ coords, int total) {
  int g = blockIdx.x * 256 + threadIdx.x;
  if (g >= total) return;
  int p   = g & 31;          // pair index 0..31
  int rem = g >> 5;
  int h   = rem % heads;
  int n   = rem / heads;
  float v0 = src[(size_t)n * ld_src + h * 64 + 2 * p];
  float v1 = src[(size_t)n * ld_src + h * 64 + 2 * p + 1];
  float cx = (float)coords[n * 2]     * 1e-5f;
  float cy = (float)coords[n * 2 + 1] * 1e-5f;
  int   j  = (p < 16) ? p : (p - 16);
  float invf = __powf(10000.0f, -(float)j * (1.0f / 16.0f));
  float f  = ((p < 16) ? cx : cy) * invf;
  float cs = __cosf(f), sn = __sinf(f);
  dst[(size_t)n * ld_dst + h * 64 + 2 * p]     = f2b(v0 * cs - v1 * sn);
  dst[(size_t)n * ld_dst + h * 64 + 2 * p + 1] = f2b(v1 * cs + v0 * sn);
}

// ---------------- fused indexer scores + top-k ----------------
// block: 256 threads (8 waves) handles one (b, 16-query tile).
// LDS: bf16 scores [16][4096] (128KB) + per-row counters.
__global__ __launch_bounds__(256)
void scores_topk(const bf16_t* __restrict__ Qi, const bf16_t* __restrict__ Ki,
                 const float* __restrict__ head_w, int* __restrict__ idx_out) {
  extern __shared__ char smem[];
  bf16_t* sc   = (bf16_t*)smem;                       // [16][TSEQ]
  int* cnt_gt  = (int*)(smem + 16 * TSEQ * 2);
  int* cnt_tie = cnt_gt + 16;

  int b    = blockIdx.y;
  int t0   = blockIdx.x << 4;
  int tid  = threadIdx.x;
  int lane = tid & 31;
  int wave = tid >> 5;

  float hw[HI_N];
#pragma unroll
  for (int h = 0; h < HI_N; ++h) hw[h] = head_w[h];

  const bf16_t* Qbase = Qi + (size_t)(b * TSEQ + t0) * 256;
  const bf16_t* Kbase = Ki + (size_t)b * TSEQ * 256;

  v16bf afr[HI_N][2];
#pragma unroll
  for (int h = 0; h < HI_N; ++h)
#pragma unroll
    for (int kc = 0; kc < 2; ++kc)
      afr[h][kc] = load_a(Qbase, 256, 0, h * 64 + kc * 32, lane);

  for (int st = wave; st < TSEQ / 16; st += 8) {
    int s0 = st << 4;
    float tot[8] = {0, 0, 0, 0, 0, 0, 0, 0};
#pragma unroll
    for (int h = 0; h < HI_N; ++h) {
      v8f ch = {};
#pragma unroll
      for (int kc = 0; kc < 2; ++kc) {
        v16bf bf = load_b(Kbase, 256, s0, h * 64 + kc * 32, lane);
        ch = __builtin_amdgcn_wmma_f32_16x16x32_bf16(false, afr[h][kc], false, bf,
                                                     (short)0, ch, false, false);
      }
#pragma unroll
      for (int r = 0; r < 8; ++r) tot[r] += hw[h] * fmaxf(ch[r], 0.0f);
    }
    int nn = s0 + (lane & 15);
    int mb = (lane >> 4) << 3;
#pragma unroll
    for (int r = 0; r < 8; ++r) sc[(mb + r) * TSEQ + nn] = f2b(tot[r]);
  }
  __syncthreads();

  // per-row radix/bisection select: 16 lanes per row (half-wave), wave-synchronous
  int row = (wave << 1) + (lane >> 4);
  int sub = lane & 15;
  const bf16_t* srow = sc + row * TSEQ;

  unsigned prefix = 0;
  for (int bit = 15; bit >= 0; --bit) {
    unsigned thr = prefix | (1u << bit);
    int c = 0;
    for (int j = sub; j < TSEQ; j += 16) c += (mkey16(srow[j]) >= thr) ? 1 : 0;
    c += __shfl_xor(c, 1);
    c += __shfl_xor(c, 2);
    c += __shfl_xor(c, 4);
    c += __shfl_xor(c, 8);
    if (c >= TOPKN) prefix = thr;
  }

  if (sub == 0) { cnt_gt[row] = 0; cnt_tie[row] = 0; }
  int* orow = idx_out + (size_t)(b * TSEQ + t0 + row) * TOPKN;
  for (int j = sub; j < TSEQ; j += 16) {
    if (mkey16(srow[j]) > prefix) {
      int p = atomicAdd(&cnt_gt[row], 1);
      orow[p] = j;
    }
  }
  int base = cnt_gt[row];
  for (int j = sub; j < TSEQ; j += 16) {
    if (mkey16(srow[j]) == prefix) {
      int p = atomicAdd(&cnt_tie[row], 1);
      if (base + p < TOPKN) orow[base + p] = j;
    }
  }
}

// ---------------- sparse attention per (b,t) ----------------
// block: 256 threads. LDS: Ksel 32KB, Vsel^T 32KB, q 2KB, logits 16KB, probs 8KB.
__global__ __launch_bounds__(256)
void attn_kernel(const bf16_t* __restrict__ qb, const bf16_t* __restrict__ kb,
                 const bf16_t* __restrict__ vb, const int* __restrict__ idx,
                 bf16_t* __restrict__ att) {
  extern __shared__ char smem[];
  bf16_t* k_s = (bf16_t*)smem;             // [256][64]
  bf16_t* v_t = (bf16_t*)(smem + 32768);   // [64][256] transposed
  bf16_t* q_s = (bf16_t*)(smem + 65536);   // [16][64] (rows 12..15 zero)
  float*  l_s = (float*)(smem + 67584);    // [16][256]
  bf16_t* p_s = (bf16_t*)(smem + 83968);   // [16][256]

  int n    = blockIdx.x;
  int b    = n >> 12;                      // T = 4096
  int tid  = threadIdx.x;
  int lane = tid & 31;
  int wave = tid >> 5;

  // gather K/V rows for this query's selected set
  int j = tid;
  int s = idx[(size_t)n * TOPKN + j];
  const bf16_t* krow = kb + (size_t)(b * TSEQ + s) * 64;
  const bf16_t* vrow = vb + (size_t)(b * TSEQ + s) * 64;
#pragma unroll
  for (int d = 0; d < 64; d += 8)
    *(uint4*)(k_s + j * 64 + d) = *(const uint4*)(krow + d);
#pragma unroll
  for (int d = 0; d < 64; ++d)
    v_t[d * 256 + j] = vrow[d];
  // stage q (12 heads x 64 = 768 elems), zero rows 12..15
  const bf16_t* qrow = qb + (size_t)n * DMODEL;
  if (tid < 96)       *(uint4*)(q_s + tid * 8) = *(const uint4*)(qrow + tid * 8);
  else if (tid < 128) *(uint4*)(q_s + tid * 8) = make_uint4(0u, 0u, 0u, 0u);
  __syncthreads();

  // logits = q * Ksel^T * (1/sqrt(64)) : M=16(heads), N=256, K=64
#pragma unroll
  for (int ti = 0; ti < 2; ++ti) {
    int s0 = (wave + ti * 8) << 4;
    v8f c = {};
#pragma unroll
    for (int kc = 0; kc < 2; ++kc) {
      v16bf a  = load_a(q_s, 64, 0, kc * 32, lane);
      v16bf bf = load_b(k_s, 64, s0, kc * 32, lane);
      c = __builtin_amdgcn_wmma_f32_16x16x32_bf16(false, a, false, bf, (short)0, c, false, false);
    }
    int nn = s0 + (lane & 15);
    int mb = (lane >> 4) << 3;
#pragma unroll
    for (int r = 0; r < 8; ++r) l_s[(mb + r) * 256 + nn] = c[r] * 0.125f;
  }
  __syncthreads();

  // softmax over 256 per head row (16 lanes per row, wave-synchronous)
  int row = (wave << 1) + (lane >> 4);
  int sub = lane & 15;
  if (row < NHEADS) {
    float mx = -3.4e38f;
    for (int jj = sub; jj < 256; jj += 16) mx = fmaxf(mx, l_s[row * 256 + jj]);
    mx = fmaxf(mx, __shfl_xor(mx, 1));
    mx = fmaxf(mx, __shfl_xor(mx, 2));
    mx = fmaxf(mx, __shfl_xor(mx, 4));
    mx = fmaxf(mx, __shfl_xor(mx, 8));
    float sum = 0.f;
    for (int jj = sub; jj < 256; jj += 16) {
      float e = __expf(l_s[row * 256 + jj] - mx);
      l_s[row * 256 + jj] = e;
      sum += e;
    }
    sum += __shfl_xor(sum, 1);
    sum += __shfl_xor(sum, 2);
    sum += __shfl_xor(sum, 4);
    sum += __shfl_xor(sum, 8);
    float inv = 1.0f / sum;
    for (int jj = sub; jj < 256; jj += 16)
      p_s[row * 256 + jj] = f2b(l_s[row * 256 + jj] * inv);
  } else {
    for (int jj = sub; jj < 256; jj += 16) p_s[row * 256 + jj] = 0;
  }
  __syncthreads();

  // out = probs * Vsel : M=16, N=64, K=256 (waves 0..3, one 16-col tile each)
  if (wave < 4) {
    int d0 = wave << 4;
    v8f c = {};
#pragma unroll
    for (int kc = 0; kc < 8; ++kc) {
      v16bf a  = load_a(p_s, 256, 0, kc * 32, lane);
      v16bf bf = load_b(v_t, 256, d0, kc * 32, lane);
      c = __builtin_amdgcn_wmma_f32_16x16x32_bf16(false, a, false, bf, (short)0, c, false, false);
    }
    int nn = lane & 15;
    int mb = (lane >> 4) << 3;
#pragma unroll
    for (int r = 0; r < 8; ++r) {
      int m = mb + r;
      if (m < NHEADS)
        att[(size_t)n * DMODEL + m * 64 + d0 + nn] = f2b(c[r]);
    }
  }
}

// ---------------- host launch ----------------
extern "C" void kernel_launch(void* const* d_in, const int* in_sizes, int n_in,
                              void* d_out, int out_size, void* d_ws, size_t ws_size,
                              hipStream_t stream) {
  const float* x      = (const float*)d_in[0];
  const int*   coords = (const int*)d_in[1];
  const float* Wq     = (const float*)d_in[2];
  const float* bq     = (const float*)d_in[3];
  const float* Wkv    = (const float*)d_in[4];
  const float* bkv    = (const float*)d_in[5];
  const float* Wo     = (const float*)d_in[6];
  const float* bo     = (const float*)d_in[7];
  const float* Wiq    = (const float*)d_in[8];
  const float* biq    = (const float*)d_in[9];
  const float* Wik    = (const float*)d_in[10];
  const float* bik    = (const float*)d_in[11];
  const float* head_w = (const float*)d_in[12];
  float* out = (float*)d_out;

  char* ws = (char*)d_ws;
  size_t off = 0;
  auto alloc = [&](size_t bytes) {
    char* p = ws + off;
    off = (off + bytes + 255) & ~(size_t)255;
    return p;
  };

  bf16_t* xb     = (bf16_t*)alloc((size_t)NTOK * DMODEL * 2);
  bf16_t* Wqb    = (bf16_t*)alloc((size_t)DMODEL * DMODEL * 2);
  bf16_t* Wkvb   = (bf16_t*)alloc((size_t)128 * DMODEL * 2);
  bf16_t* Wiqb   = (bf16_t*)alloc((size_t)256 * DMODEL * 2);
  bf16_t* Wikb   = (bf16_t*)alloc((size_t)256 * DMODEL * 2);
  bf16_t* Wob    = (bf16_t*)alloc((size_t)DMODEL * DMODEL * 2);
  float*  q_f32  = (float*)alloc((size_t)NTOK * DMODEL * 4);
  float*  kv_f32 = (float*)alloc((size_t)NTOK * 128 * 4);
  bf16_t* q_b    = (bf16_t*)alloc((size_t)NTOK * DMODEL * 2);
  bf16_t* k_b    = (bf16_t*)alloc((size_t)NTOK * 64 * 2);
  bf16_t* v_b    = (bf16_t*)alloc((size_t)NTOK * 64 * 2);
  bf16_t* Qi_b   = (bf16_t*)alloc((size_t)NTOK * 256 * 2);
  bf16_t* Ki_b   = (bf16_t*)alloc((size_t)NTOK * 256 * 2);
  int*    idxb   = (int*)alloc((size_t)NTOK * TOPKN * 4);
  bf16_t* att_b  = (bf16_t*)alloc((size_t)NTOK * DMODEL * 2);

  auto cdiv = [](int a, int b) { return (a + b - 1) / b; };

  // bf16 conversion of activations + weights
  cvt_f2b<<<cdiv(NTOK * DMODEL, 256), 256, 0, stream>>>(x, xb, NTOK * DMODEL);
  cvt_f2b<<<cdiv(DMODEL * DMODEL, 256), 256, 0, stream>>>(Wq, Wqb, DMODEL * DMODEL);
  cvt_f2b<<<cdiv(128 * DMODEL, 256), 256, 0, stream>>>(Wkv, Wkvb, 128 * DMODEL);
  cvt_f2b<<<cdiv(256 * DMODEL, 256), 256, 0, stream>>>(Wiq, Wiqb, 256 * DMODEL);
  cvt_f2b<<<cdiv(256 * DMODEL, 256), 256, 0, stream>>>(Wik, Wikb, 256 * DMODEL);
  cvt_f2b<<<cdiv(DMODEL * DMODEL, 256), 256, 0, stream>>>(Wo, Wob, DMODEL * DMODEL);

  // projections (WMMA, 16x64 strip per wave, 4 waves per block)
  gemm_bf16<<<(NTOK / 16) * (DMODEL / 64) / 4, 128, 0, stream>>>(xb, Wqb, bq, q_f32, nullptr,
                                                                 NTOK, DMODEL, DMODEL);
  gemm_bf16<<<(NTOK / 16) * (128 / 64) / 4, 128, 0, stream>>>(xb, Wkvb, bkv, kv_f32, nullptr,
                                                              NTOK, 128, DMODEL);
  gemm_bf16<<<(NTOK / 16) * (256 / 64) / 4, 128, 0, stream>>>(xb, Wiqb, biq, nullptr, Qi_b,
                                                              NTOK, 256, DMODEL);
  gemm_bf16<<<(NTOK / 16) * (256 / 64) / 4, 128, 0, stream>>>(xb, Wikb, bik, nullptr, Ki_b,
                                                              NTOK, 256, DMODEL);

  // RoPE(q), RoPE(k), pack v
  rope_kernel<<<cdiv(NTOK * NHEADS * 32, 256), 256, 0, stream>>>(q_f32, DMODEL, q_b, DMODEL,
                                                                 NHEADS, coords,
                                                                 NTOK * NHEADS * 32);
  rope_kernel<<<cdiv(NTOK * 32, 256), 256, 0, stream>>>(kv_f32, 128, k_b, 64, 1, coords,
                                                        NTOK * 32);
  cvt_v<<<cdiv(NTOK * 64, 256), 256, 0, stream>>>(kv_f32, v_b, NTOK * 64);

  // fused indexer scores + top-k (bf16 scores in LDS: 16*4096*2 + counters)
  {
    dim3 g(TSEQ / 16, BATCH);
    size_t sm = (size_t)16 * TSEQ * 2 + 32 * sizeof(int);
    scores_topk<<<g, 256, sm, stream>>>(Qi_b, Ki_b, head_w, idxb);
  }

  // sparse attention per (b,t)
  {
    size_t sm = 32768 + 32768 + 2048 + 16384 + 8192; // 92160 bytes
    attn_kernel<<<NTOK, 256, sm, stream>>>(q_b, k_b, v_b, idxb, att_b);
  }

  // output projection -> d_out (f32)
  gemm_bf16<<<(NTOK / 16) * (DMODEL / 64) / 4, 128, 0, stream>>>(att_b, Wob, bo, out, nullptr,
                                                                 NTOK, DMODEL, DMODEL);

  (void)in_sizes; (void)n_in; (void)out_size; (void)ws_size;
}
